// Attention_17231408791694
// MI455X (gfx1250) — matrix operand order repaired and checked by reading, not compiled, not run on hardware
//
#include <hip/hip_runtime.h>
#include <hip/hip_fp16.h>
#include <stdint.h>

#define DIM    512
#define INNER  64
#define HEADS  8
#define BATCH  2
#define SEQ    4096

typedef __attribute__((ext_vector_type(16))) _Float16 v16h;
typedef __attribute__((ext_vector_type(8)))  float    v8f;

union HFrag { v16h h; uint32_t u[8]; };

// ---------------------------------------------------------------------------
// A-fragment (16x32 f16, row-major source, stride lda elements).
// lane&15 = row; lane>>4 selects K-half; VGPR g holds packed K pair.
// ---------------------------------------------------------------------------
__device__ __forceinline__ v16h load_tileA(const _Float16* base, int lda, int lane) {
    const int r     = lane & 15;
    const int khalf = (lane >> 4) << 4;
    const _Float16* rowp = base + (size_t)r * lda;
    HFrag f;
#pragma unroll
    for (int g = 0; g < 8; ++g) {
        const int k = khalf + (g << 1);
        f.u[g] = *(const uint32_t*)(rowp + k);
    }
    return f.h;
}

// ---------------------------------------------------------------------------
// B-fragment (32x16 f16). lane&15 = column n, element e -> k = (lane>>4)*16+e.
// Source stored transposed: B[k][n] = base[n*ldb + k] -> 16 contiguous f16/lane.
// ---------------------------------------------------------------------------
__device__ __forceinline__ v16h load_tileB(const _Float16* base, int ldb, int lane) {
    const int n  = lane & 15;
    const int k0 = (lane >> 4) << 4;
    const _Float16* p = base + (size_t)n * ldb + k0;
    HFrag f;
#pragma unroll
    for (int g = 0; g < 8; ++g) f.u[g] = *(const uint32_t*)(p + (g << 1));
    return f.h;
}

__device__ __forceinline__ v8f wmma_f16(v16h a, v16h b, v8f c) {
    return __builtin_amdgcn_wmma_f32_16x16x32_f16(
        false, a, false, b, (short)0, c, false, false);
}

// Low 32 bits of a generic pointer to LDS == wave-relative LDS byte address
__device__ __forceinline__ uint32_t lds_addr32(const void* p) {
    return (uint32_t)(uintptr_t)p;
}

#define WAIT_ASYNC(n) asm volatile("s_wait_asynccnt %0" :: "i"(n) : "memory")

// One 16-byte async global->LDS copy (ASYNCcnt-tracked)
__device__ __forceinline__ void async_b128(uint32_t lds, uint64_t gaddr) {
    asm volatile("global_load_async_to_lds_b128 %0, %1, off"
                 :: "v"(lds), "v"(gaddr) : "memory");
}

// Stage one 32-key K block (4KB contiguous) and one 64e x 32key V block (4KB,
// 64B rows) into LDS: 16 async b128 issues per lane.
__device__ __forceinline__ void stage_kv(const _Float16* Kg, const _Float16* Vtg,
                                         uint32_t kl, uint32_t vl, int lane) {
    const uint64_t gk = (uint64_t)(uintptr_t)Kg;
#pragma unroll
    for (int c = 0; c < 8; ++c) {
        const uint32_t off = (uint32_t)((c * 32 + lane) * 16);
        async_b128(kl + off, gk + off);
    }
#pragma unroll
    for (int rr = 0; rr < 2; ++rr) {
        const int e = lane * 2 + rr;
        const uint64_t gv = (uint64_t)(uintptr_t)(Vtg + (size_t)e * SEQ);
        const uint32_t vb = vl + (uint32_t)e * 64;
#pragma unroll
        for (int c = 0; c < 4; ++c) async_b128(vb + c * 16, gv + c * 16);
    }
}

// ------------------------------ conversions --------------------------------
__global__ void cvt_f32_f16(const float* __restrict__ in, _Float16* __restrict__ out, int n) {
    const int i = blockIdx.x * blockDim.x + threadIdx.x;
    if (i < n) out[i] = (_Float16)in[i];
}

// W [H,D,E] f32 -> Wt [H,E,D] f16
__global__ void cvt_w_hde_to_hed(const float* __restrict__ in, _Float16* __restrict__ out) {
    const int i = blockIdx.x * blockDim.x + threadIdx.x;
    if (i >= HEADS * DIM * INNER) return;
    const int h = i / (DIM * INNER);
    const int r = i - h * (DIM * INNER);
    const int d = r / INNER;
    const int e = r - d * INNER;
    out[((size_t)h * INNER + e) * DIM + d] = (_Float16)in[i];
}

// Wo [K][N] f32 -> Wot [N][K] f16
__global__ void cvt_wo_T(const float* __restrict__ in, _Float16* __restrict__ out) {
    const int i = blockIdx.x * blockDim.x + threadIdx.x;
    if (i >= DIM * DIM) return;
    const int k = i / DIM;
    const int n = i - k * DIM;
    out[(size_t)n * DIM + k] = (_Float16)in[i];
}

// ------------------------------ QKV projection -----------------------------
// One wave per (32-row tile, head, {Q,K,V}): 32x64 tile, register-double-
// buffered K loop (last iteration peeled): 8 WMMA per 6 fragment loads.
__global__ void __launch_bounds__(128, 1)
qkv_kernel(const _Float16* __restrict__ xh,
           const _Float16* __restrict__ Wqt,
           const _Float16* __restrict__ Wkt,
           const _Float16* __restrict__ Wvt,
           const float* __restrict__ bq,
           const float* __restrict__ bk,
           const float* __restrict__ bv,
           _Float16* __restrict__ Q,
           _Float16* __restrict__ K,
           _Float16* __restrict__ Vt) {
    const int lane = threadIdx.x & 31;
    const int wave = blockIdx.x * (blockDim.x >> 5) + (threadIdx.x >> 5);
    const int MT32 = (BATCH * SEQ) / 32;   // 256
    const int mtile = wave % MT32;
    const int rest  = wave / MT32;
    const int h     = rest & (HEADS - 1);
    const int mat   = rest >> 3;           // 0=Q,1=K,2=V

    const _Float16* Wt  = (mat == 0) ? Wqt : (mat == 1) ? Wkt : Wvt;
    const float*    bia = (mat == 0) ? bq  : (mat == 1) ? bk  : bv;

    v8f acc[2][4];
#pragma unroll
    for (int i = 0; i < 2; ++i)
#pragma unroll
        for (int t = 0; t < 4; ++t) acc[i][t] = v8f{};

    const _Float16* abase = xh + (size_t)mtile * 32 * DIM;
    const _Float16* wbase = Wt + (size_t)h * INNER * DIM;

    v16h a0 = load_tileA(abase, DIM, lane);
    v16h a1 = load_tileA(abase + 16 * DIM, DIM, lane);
    v16h b0 = load_tileB(wbase + 0 * 16 * DIM, DIM, lane);
    v16h b1 = load_tileB(wbase + 1 * 16 * DIM, DIM, lane);
    v16h b2 = load_tileB(wbase + 2 * 16 * DIM, DIM, lane);
    v16h b3 = load_tileB(wbase + 3 * 16 * DIM, DIM, lane);

    for (int kb = 0; kb < DIM - 32; kb += 32) {
        const int nk = kb + 32;
        const v16h na0 = load_tileA(abase + nk, DIM, lane);
        const v16h na1 = load_tileA(abase + 16 * DIM + nk, DIM, lane);
        const v16h nb0 = load_tileB(wbase + 0 * 16 * DIM + nk, DIM, lane);
        const v16h nb1 = load_tileB(wbase + 1 * 16 * DIM + nk, DIM, lane);
        const v16h nb2 = load_tileB(wbase + 2 * 16 * DIM + nk, DIM, lane);
        const v16h nb3 = load_tileB(wbase + 3 * 16 * DIM + nk, DIM, lane);

        acc[0][0] = wmma_f16(a0, b0, acc[0][0]);
        acc[1][0] = wmma_f16(a1, b0, acc[1][0]);
        acc[0][1] = wmma_f16(a0, b1, acc[0][1]);
        acc[1][1] = wmma_f16(a1, b1, acc[1][1]);
        acc[0][2] = wmma_f16(a0, b2, acc[0][2]);
        acc[1][2] = wmma_f16(a1, b2, acc[1][2]);
        acc[0][3] = wmma_f16(a0, b3, acc[0][3]);
        acc[1][3] = wmma_f16(a1, b3, acc[1][3]);

        a0 = na0; a1 = na1; b0 = nb0; b1 = nb1; b2 = nb2; b3 = nb3;
    }
    // peeled last K-slice
    acc[0][0] = wmma_f16(a0, b0, acc[0][0]);
    acc[1][0] = wmma_f16(a1, b0, acc[1][0]);
    acc[0][1] = wmma_f16(a0, b1, acc[0][1]);
    acc[1][1] = wmma_f16(a1, b1, acc[1][1]);
    acc[0][2] = wmma_f16(a0, b2, acc[0][2]);
    acc[1][2] = wmma_f16(a1, b2, acc[1][2]);
    acc[0][3] = wmma_f16(a0, b3, acc[0][3]);
    acc[1][3] = wmma_f16(a1, b3, acc[1][3]);

    const int col0  = lane & 15;
    const int rhalf = (lane >> 4) << 3;
#pragma unroll
    for (int t = 0; t < 4; ++t) {
        const int e = t * 16 + col0;
        const float bval = bia[h * INNER + e];
#pragma unroll
        for (int i = 0; i < 2; ++i) {
#pragma unroll
            for (int g = 0; g < 8; ++g) {
                const int m = mtile * 32 + i * 16 + rhalf + g;
                const int bb = m >> 12;
                const int s  = m & (SEQ - 1);
                const float v = acc[i][t][g] + bval;
                if (mat == 2) {
                    Vt[((size_t)(bb * HEADS + h) * INNER + e) * SEQ + s] = (_Float16)v;
                } else {
                    _Float16* dst = (mat == 0) ? Q : K;
                    dst[((size_t)(bb * HEADS + h) * SEQ + s) * INNER + e] = (_Float16)v;
                }
            }
        }
    }
}

// ------------------------------ flash attention ----------------------------
// One wave per (b,h, 16-query tile); 2 waves/block. K/V key-blocks staged into
// LDS via async global->LDS copies, double-buffered (ASYNCcnt). Row-sums of P
// computed with a WMMA against an all-ones fragment.
__global__ void __launch_bounds__(64, 1)
attn_kernel(const _Float16* __restrict__ Q,
            const _Float16* __restrict__ K,
            const _Float16* __restrict__ Vt,
            _Float16* __restrict__ Y) {
    __shared__ _Float16 Kbuf[2][2][32 * 64];   // [wave][buf][key][e]
    __shared__ _Float16 Vbuf[2][2][64 * 32];   // [wave][buf][e][key]
    __shared__ _Float16 Pbuf[2][16 * 32];

    const int lane  = threadIdx.x & 31;
    const int wslot = threadIdx.x >> 5;
    const int wave  = blockIdx.x * 2 + wslot;
    const int ST    = SEQ / 16;                // 256
    const int stile = wave % ST;
    const int bh    = wave / ST;

    const _Float16* qbase = Q + ((size_t)bh * SEQ + (size_t)stile * 16) * INNER;
    const v16h aq0 = load_tileA(qbase,      INNER, lane);   // e = 0..31
    const v16h aq1 = load_tileA(qbase + 32, INNER, lane);   // e = 32..63

    HFrag onesf;
#pragma unroll
    for (int g = 0; g < 8; ++g) onesf.u[g] = 0x3C003C00u;   // f16 1.0 pairs

    v8f o[4];
#pragma unroll
    for (int t = 0; t < 4; ++t) o[t] = v8f{};
    float m[8], l[8];
#pragma unroll
    for (int g = 0; g < 8; ++g) { m[g] = -1e30f; l[g] = 0.f; }

    const int col0  = lane & 15;
    const int rhalf = (lane >> 4) << 3;
    _Float16* Pl = Pbuf[wslot];
    const float scale = 0.125f;                // 1/sqrt(64)

    const _Float16* Kg0  = K  + (size_t)bh * SEQ * INNER;
    const _Float16* Vtg0 = Vt + (size_t)bh * INNER * SEQ;
    const uint32_t kl[2] = { lds_addr32(Kbuf[wslot][0]), lds_addr32(Kbuf[wslot][1]) };
    const uint32_t vl[2] = { lds_addr32(Vbuf[wslot][0]), lds_addr32(Vbuf[wslot][1]) };

    stage_kv(Kg0, Vtg0, kl[0], vl[0], lane);   // prologue: block 0 -> buf 0
    int cur = 0;

    for (int kb = 0; kb < SEQ; kb += 32) {
        if (kb + 32 < SEQ) {                   // issue next block, then wait for current
            stage_kv(Kg0 + (size_t)(kb + 32) * INNER, Vtg0 + (kb + 32),
                     kl[cur ^ 1], vl[cur ^ 1], lane);
            WAIT_ASYNC(16);
        } else {
            WAIT_ASYNC(0);
        }
        const _Float16* Kl = Kbuf[wslot][cur];
        const _Float16* Vl = Vbuf[wslot][cur];

        v8f s0 = v8f{}, s1 = v8f{};
        s0 = wmma_f16(aq0, load_tileB(Kl,                   INNER, lane), s0);
        s0 = wmma_f16(aq1, load_tileB(Kl + 32,              INNER, lane), s0);
        s1 = wmma_f16(aq0, load_tileB(Kl + 16 * INNER,      INNER, lane), s1);
        s1 = wmma_f16(aq1, load_tileB(Kl + 16 * INNER + 32, INNER, lane), s1);

        float p0[8], p1[8], alpha[8];
#pragma unroll
        for (int g = 0; g < 8; ++g) {
            const float x0 = s0[g] * scale;
            const float x1 = s1[g] * scale;
            float mx = fmaxf(x0, x1);
#pragma unroll
            for (int off = 8; off >= 1; off >>= 1)
                mx = fmaxf(mx, __shfl_xor(mx, off, 16));    // row max over 16 cols
            const float mn = fmaxf(m[g], mx);
            alpha[g] = __expf(m[g] - mn);
            m[g] = mn;
            p0[g] = __expf(x0 - mn);
            p1[g] = __expf(x1 - mn);
        }
#pragma unroll
        for (int t = 0; t < 4; ++t)
#pragma unroll
            for (int g = 0; g < 8; ++g) o[t][g] *= alpha[g];

        // C-layout -> A-layout reshape of P through per-wave LDS tile
#pragma unroll
        for (int g = 0; g < 8; ++g) {
            const int r = rhalf + g;
            Pl[r * 32 + col0]      = (_Float16)p0[g];
            Pl[r * 32 + 16 + col0] = (_Float16)p1[g];
        }
        __syncthreads();
        const v16h ap = load_tileA(Pl, 32, lane);

        // row-sums of P via WMMA against ones: result col-invariant per row
        const v8f rs = wmma_f16(ap, onesf.h, v8f{});
#pragma unroll
        for (int g = 0; g < 8; ++g) l[g] = l[g] * alpha[g] + rs[g];

#pragma unroll
        for (int t = 0; t < 4; ++t)
            o[t] = wmma_f16(ap, load_tileB(Vl + (size_t)(t * 16) * 32, 32, lane), o[t]);

        cur ^= 1;
    }

    const int b = bh / HEADS;
    const int h = bh - b * HEADS;
#pragma unroll
    for (int t = 0; t < 4; ++t) {
#pragma unroll
        for (int g = 0; g < 8; ++g) {
            const int s = stile * 16 + rhalf + g;
            const float v = o[t][g] / l[g];
            Y[(size_t)(b * SEQ + s) * DIM + h * INNER + t * 16 + col0] = (_Float16)v;
        }
    }
}

// ------------------------------ output projection --------------------------
// One wave per 32x32 tile, register-double-buffered (last iteration peeled).
__global__ void __launch_bounds__(128, 1)
proj_kernel(const _Float16* __restrict__ Yh,
            const _Float16* __restrict__ Wot,
            const float* __restrict__ bo,
            float* __restrict__ out) {
    const int lane = threadIdx.x & 31;
    const int wave = blockIdx.x * (blockDim.x >> 5) + (threadIdx.x >> 5);
    const int MT32 = (BATCH * SEQ) / 32;   // 256
    const int mtile = wave % MT32;
    const int nt    = wave / MT32;         // 0..15 (32-col tiles)

    v8f acc[2][2];
#pragma unroll
    for (int i = 0; i < 2; ++i)
#pragma unroll
        for (int j = 0; j < 2; ++j) acc[i][j] = v8f{};

    const _Float16* abase = Yh  + (size_t)mtile * 32 * DIM;
    const _Float16* bbase = Wot + (size_t)nt    * 32 * DIM;

    v16h a0 = load_tileA(abase, DIM, lane);
    v16h a1 = load_tileA(abase + 16 * DIM, DIM, lane);
    v16h b0 = load_tileB(bbase, DIM, lane);
    v16h b1 = load_tileB(bbase + 16 * DIM, DIM, lane);

    for (int kb = 0; kb < DIM - 32; kb += 32) {
        const int nk = kb + 32;
        const v16h na0 = load_tileA(abase + nk, DIM, lane);
        const v16h na1 = load_tileA(abase + 16 * DIM + nk, DIM, lane);
        const v16h nb0 = load_tileB(bbase + nk, DIM, lane);
        const v16h nb1 = load_tileB(bbase + 16 * DIM + nk, DIM, lane);

        acc[0][0] = wmma_f16(a0, b0, acc[0][0]);
        acc[0][1] = wmma_f16(a0, b1, acc[0][1]);
        acc[1][0] = wmma_f16(a1, b0, acc[1][0]);
        acc[1][1] = wmma_f16(a1, b1, acc[1][1]);

        a0 = na0; a1 = na1; b0 = nb0; b1 = nb1;
    }
    acc[0][0] = wmma_f16(a0, b0, acc[0][0]);
    acc[0][1] = wmma_f16(a0, b1, acc[0][1]);
    acc[1][0] = wmma_f16(a1, b0, acc[1][0]);
    acc[1][1] = wmma_f16(a1, b1, acc[1][1]);

    const int col0  = lane & 15;
    const int rhalf = (lane >> 4) << 3;
#pragma unroll
    for (int j = 0; j < 2; ++j) {
        const int col = nt * 32 + j * 16 + col0;
        const float bv = bo[col];
#pragma unroll
        for (int i = 0; i < 2; ++i) {
#pragma unroll
            for (int g = 0; g < 8; ++g) {
                const int mrow = mtile * 32 + i * 16 + rhalf + g;
                out[(size_t)mrow * DIM + col] = acc[i][j][g] + bv;
            }
        }
    }
}

// ------------------------------ launcher -----------------------------------
extern "C" void kernel_launch(void* const* d_in, const int* in_sizes, int n_in,
                              void* d_out, int out_size, void* d_ws, size_t ws_size,
                              hipStream_t stream) {
    (void)in_sizes; (void)n_in; (void)out_size; (void)ws_size;
    const float* x  = (const float*)d_in[0];
    const float* Wq = (const float*)d_in[1];
    const float* bq = (const float*)d_in[2];
    const float* Wk = (const float*)d_in[3];
    const float* bk = (const float*)d_in[4];
    const float* Wv = (const float*)d_in[5];
    const float* bv = (const float*)d_in[6];
    const float* Wo = (const float*)d_in[7];
    const float* bo = (const float*)d_in[8];
    float* out = (float*)d_out;

    char* ws = (char*)d_ws;
    size_t off = 0;
    auto alloc = [&](size_t bytes) -> char* {
        char* p = ws + off;
        off = (off + bytes + 255) & ~(size_t)255;
        return p;
    };
    _Float16* xh  = (_Float16*)alloc((size_t)BATCH * SEQ * DIM * 2);
    _Float16* Wqt = (_Float16*)alloc((size_t)HEADS * INNER * DIM * 2);
    _Float16* Wkt = (_Float16*)alloc((size_t)HEADS * INNER * DIM * 2);
    _Float16* Wvt = (_Float16*)alloc((size_t)HEADS * INNER * DIM * 2);
    _Float16* Wot = (_Float16*)alloc((size_t)DIM * DIM * 2);
    _Float16* Qb  = (_Float16*)alloc((size_t)BATCH * HEADS * SEQ * INNER * 2);
    _Float16* Kb  = (_Float16*)alloc((size_t)BATCH * HEADS * SEQ * INNER * 2);
    _Float16* Vtb = (_Float16*)alloc((size_t)BATCH * HEADS * INNER * SEQ * 2);
    _Float16* Yb  = (_Float16*)alloc((size_t)BATCH * SEQ * DIM * 2);

    {
        const int n = BATCH * SEQ * DIM;
        cvt_f32_f16<<<(n + 255) / 256, 256, 0, stream>>>(x, xh, n);
    }
    const int nw = HEADS * DIM * INNER;
    cvt_w_hde_to_hed<<<(nw + 255) / 256, 256, 0, stream>>>(Wq, Wqt);
    cvt_w_hde_to_hed<<<(nw + 255) / 256, 256, 0, stream>>>(Wk, Wkt);
    cvt_w_hde_to_hed<<<(nw + 255) / 256, 256, 0, stream>>>(Wv, Wvt);
    cvt_wo_T<<<(DIM * DIM + 255) / 256, 256, 0, stream>>>(Wo, Wot);

    // QKV: 256 mtiles * 8 heads * 3 mats = 6144 waves, 4 waves/block
    qkv_kernel<<<6144 / 4, 128, 0, stream>>>(xh, Wqt, Wkt, Wvt, bq, bk, bv, Qb, Kb, Vtb);

    // attention: 2*8*256 = 4096 waves, 2 waves/block (LDS double buffers)
    attn_kernel<<<4096 / 2, 64, 0, stream>>>(Qb, Kb, Vtb, Yb);

    // projection: 256 mtiles * 16 ntiles = 4096 waves
    proj_kernel<<<4096 / 4, 128, 0, stream>>>(Yb, Wot, bo, out);
}
